// GraphConvolution_17824114278401
// MI455X (gfx1250) — matrix operand order repaired and checked by reading, not compiled, not run on hardware
//
#include <hip/hip_runtime.h>
#include <cstdint>
#include <cstddef>

// ---------------------------------------------------------------------------
// GCN attention-style fused GraphConvolution for gfx1250 (wave32, WMMA bf16,
// Tensor Data Mover staging with double-buffered K/V tiles)
//   out = softmax(xn xn^T) (xt W) + b + xt     (per batch), out in [B,C,N]
// ---------------------------------------------------------------------------

typedef __attribute__((ext_vector_type(16))) __bf16   v16bf;
typedef __attribute__((ext_vector_type(8)))  float    v8f;
typedef __attribute__((ext_vector_type(4)))  unsigned su4;
typedef __attribute__((ext_vector_type(8)))  unsigned su8;

#define B_   32
#define C_   256
#define N_   2048
#define MT   128            // Q rows per workgroup (8 waves x 16)
#define KT   64             // keys per K-loop iteration
#define QS   264            // LDS row stride (elems) for Q/K tiles (TDM pad: 4 dwords / 512B)
#define VS   72             // LDS row stride (elems) for Vt / P tiles (TDM pad: 4 dwords / 128B)

// LDS layout (element offsets into the dynamic-LDS block; byte = elem*2)
#define Q_E   (MT * QS)                 // 33792
#define K_E   (KT * QS)                 // 16896
#define V_E   (C_ * VS)                 // 18432
#define KB_E  (Q_E)                     // K buffers base
#define VB_E  (Q_E + 2 * K_E)           // Vt buffers base
#define P_E   (Q_E + 2 * K_E + 2 * V_E) // P staging base
#define SMEM_BYTES ((P_E + 8 * 16 * VS) * 2)   // 227328

static __device__ __forceinline__ unsigned short f2bf(float f) {
  unsigned u = __float_as_uint(f);
  u += 0x7FFFu + ((u >> 16) & 1u);          // round-to-nearest-even
  return (unsigned short)(u >> 16);
}

union V16U { uint4 q[2]; v16bf v; };

// load 16 bf16 as two 16B chunks (possibly non-adjacent) -> WMMA operand
static __device__ __forceinline__ v16bf ld2(const unsigned short* p0,
                                            const unsigned short* p1) {
  V16U u;
  u.q[0] = *reinterpret_cast<const uint4*>(p0);
  u.q[1] = *reinterpret_cast<const uint4*>(p1);
  return u.v;
}

static __device__ __forceinline__ v8f zero8() {
  v8f z; for (int i = 0; i < 8; ++i) z[i] = 0.f; return z;
}

// reductions across the 16-lane half-group (wave32; C/D rows live per half)
static __device__ __forceinline__ float grp_max(float v) {
  v = fmaxf(v, __shfl_xor(v, 1, 32));
  v = fmaxf(v, __shfl_xor(v, 2, 32));
  v = fmaxf(v, __shfl_xor(v, 4, 32));
  v = fmaxf(v, __shfl_xor(v, 8, 32));
  return v;
}
static __device__ __forceinline__ float grp_sum(float v) {
  v += __shfl_xor(v, 1, 32);
  v += __shfl_xor(v, 2, 32);
  v += __shfl_xor(v, 4, 32);
  v += __shfl_xor(v, 8, 32);
  return v;
}

// ---------------------------------------------------------------------------
// Tensor Data Mover: 2D tile (bf16) global -> LDS with row padding.
// D# packing per CDNA5 ISA 8.3/8.4: group0 = {flags, lds_addr, global_addr,
// type=2}; group1 = {data_size=2B, pad cfg, dims, tile dims, dim0 stride}.
static __device__ __forceinline__ void tdm_load_2d(unsigned lds_byte_off,
                                                   const void* gaddr,
                                                   unsigned tensor_d0,
                                                   unsigned tensor_d1,
                                                   unsigned tile_d0,
                                                   unsigned tile_d1,
                                                   unsigned stride0,
                                                   unsigned pad_interval,
                                                   unsigned pad_amount) {
  unsigned long long ga = (unsigned long long)(uintptr_t)gaddr;
  su4 g0;
  g0[0] = 1u;                                            // count=1, user mode
  g0[1] = lds_byte_off;                                  // lds_addr
  g0[2] = (unsigned)ga;                                  // global_addr[31:0]
  g0[3] = (unsigned)((ga >> 32) & 0x1FFFFFFu) | (2u << 30); // ga[56:32], type=2
  su8 g1;
  g1[0] = (1u << 16)                                     // data_size = 2 bytes
        | (1u << 20)                                     // pad_enable
        | (pad_interval << 22) | (pad_amount << 25);
  g1[1] = (tensor_d0 & 0xFFFFu) << 16;                   // tensor_dim0[15:0]
  g1[2] = (tensor_d0 >> 16) | ((tensor_d1 & 0xFFFFu) << 16);
  g1[3] = (tensor_d1 >> 16) | (tile_d0 << 16);           // tile_dim0
  g1[4] = tile_d1;                                       // tile_dim1 (tile_dim2=0)
  g1[5] = stride0;                                       // tensor_dim0_stride lo
  g1[6] = 0;
  g1[7] = 0;
  asm volatile("tensor_load_to_lds %0, %1" : : "s"(g0), "s"(g1) : "memory");
}

// ---------------------------------------------------------------------------
// Kernel 0: W[c][o] -> Wt_bf16[o][c]
__global__ void gcn_pack_w(const float* __restrict__ W,
                           unsigned short* __restrict__ Wt) {
  int i = blockIdx.x * 256 + threadIdx.x;     // 65536 elements
  int o = i >> 8, c = i & 255;
  Wt[(size_t)o * C_ + c] = f2bf(W[(size_t)c * C_ + o]);
}

// ---------------------------------------------------------------------------
// Kernel 1: column L2-norm of x[B,C,N]; emit xt_bf16[B,N,C], xn_bf16[B,N,C]
__global__ void gcn_pack_x(const float* __restrict__ x,
                           unsigned short* __restrict__ xn,
                           unsigned short* __restrict__ xt) {
  int b = blockIdx.y;
  int n = blockIdx.x * 256 + threadIdx.x;
  const float* xp = x + (size_t)b * C_ * N_ + n;
  float ss = 0.f;
  for (int c = 0; c < C_; ++c) { float v = xp[(size_t)c * N_]; ss += v * v; }
  float inv = 1.0f / fmaxf(sqrtf(ss), 1e-12f);
  size_t o = ((size_t)b * N_ + n) * C_;
  for (int c = 0; c < C_; ++c) {
    float v = xp[(size_t)c * N_];
    xt[o + c] = f2bf(v);
    xn[o + c] = f2bf(v * inv);
  }
}

// ---------------------------------------------------------------------------
// Kernel 2: supT_bf16[B,C,N] = (xt @ W)^T   (WMMA bf16, LDS-staged transpose)
__global__ void gcn_support(const unsigned short* __restrict__ xt,
                            const unsigned short* __restrict__ Wt,
                            unsigned short* __restrict__ supT) {
  __shared__ __align__(16) unsigned short st[16][136];   // [o_local][n_local+pad]

  int b  = blockIdx.z;
  int o0 = blockIdx.x * 16;
  int tid = threadIdx.x;
  int w = tid >> 5, lane = tid & 31;
  int lm = lane & 15;
  int half = (lane < 16) ? 0 : 8;
  int n0 = blockIdx.y * 128;
  int nw = n0 + w * 16;

  const unsigned short* A  = xt + ((size_t)b * N_ + nw + lm) * C_;
  const unsigned short* Bp = Wt + (size_t)(o0 + lm) * C_;

  v8f acc = zero8();
  for (int kc = 0; kc < 8; ++kc) {
    int c0 = kc * 32;
    v16bf a  = ld2(A + c0 + half,       A + c0 + 16 + half);        // A 16x32 layout
    v16bf bm = ld2(Bp + c0 + half,  Bp + c0 + 16 + half);    // B 32x16 layout
    acc = __builtin_amdgcn_wmma_f32_16x16x32_bf16(false, a, false, bm,
                                                  (short)0, acc, false, false);
  }
  // C-layout -> LDS stage (transpose): st[o_local][n_local]
  for (int a2 = 0; a2 < 8; ++a2)
    st[lm][w * 16 + a2 + half] = f2bf(acc[a2]);
  __syncthreads();
  // coalesced 16B writes of the transposed tile
  int r = tid >> 4, ch = tid & 15;
  uint4 v = *reinterpret_cast<const uint4*>(&st[r][ch * 8]);
  *reinterpret_cast<uint4*>(supT + ((size_t)b * C_ + o0 + r) * N_ + n0 + ch * 8) = v;
}

// ---------------------------------------------------------------------------
// Kernel 3: flash attention, TDM-staged tiles, double buffered K/V.
//   out = softmax(xn xn^T) supT^T + bias + x   stored as [B,C,N]
__global__ void gcn_attn(const unsigned short* __restrict__ xn,
                         const unsigned short* __restrict__ supT,
                         const float* __restrict__ x,
                         const float* __restrict__ bias,
                         float* __restrict__ out) {
  extern __shared__ unsigned short sm[];

  int b  = blockIdx.y;
  int q0 = blockIdx.x * MT;
  int tid = threadIdx.x;
  int w = tid >> 5, lane = tid & 31;
  int lm = lane & 15;
  int half = (lane < 16) ? 0 : 8;
  int qr = w * 16;

  const unsigned short* xb  = xn  + (size_t)b * N_ * C_;
  const unsigned short* vb  = supT + (size_t)b * C_ * N_;

  // ---- prologue: TDM-load Q tile + K/V tile 0 (wave 0 issues) ----
  if (w == 0) {
    // Q: 128x256 tile of [2048 x 256] bf16, pad 4 dwords every 512B -> stride 264
    tdm_load_2d(0u, xb + (size_t)q0 * C_, C_, N_, C_, MT, C_, 6u, 3u);
    // K tile 0: 64x256
    tdm_load_2d(KB_E * 2u, xb, C_, N_, C_, KT, C_, 6u, 3u);
    // Vt tile 0: 256 rows x 64 cols of [256 x 2048], pad 4 dwords every 128B -> stride 72
    tdm_load_2d(VB_E * 2u, vb, N_, C_, KT, C_, N_, 4u, 3u);
    __builtin_amdgcn_s_wait_tensorcnt(0);
  }
  __syncthreads();

  v8f o_acc[16];
  for (int i = 0; i < 16; ++i) o_acc[i] = zero8();
  float m_st[8], l_st[8];
  for (int a = 0; a < 8; ++a) { m_st[a] = -1e30f; l_st[a] = 0.f; }

  for (int it = 0; it < N_ / KT; ++it) {
    int cur = it & 1;
    int kv_next = (it + 1) * KT;

    // ---- issue DMA for next tile into the free buffer (overlaps compute) ----
    if (w == 0 && kv_next < N_) {
      unsigned nb = (unsigned)(1 - cur);
      tdm_load_2d((KB_E + nb * K_E) * 2u, xb + (size_t)kv_next * C_,
                  C_, N_, C_, KT, C_, 6u, 3u);
      tdm_load_2d((VB_E + nb * V_E) * 2u, vb + kv_next,
                  N_, C_, KT, C_, N_, 4u, 3u);
    }

    const unsigned short* qs = sm;
    const unsigned short* ks = sm + KB_E + cur * K_E;
    const unsigned short* vt = sm + VB_E + cur * V_E;

    // ---- S = Q K^T strip (16 x 64): 32x v_wmma_f32_16x16x32_bf16 ----
    v8f s[4];
    for (int t = 0; t < 4; ++t) s[t] = zero8();
    for (int kc = 0; kc < 8; ++kc) {
      int c0 = kc * 32;
      const unsigned short* ap = qs + (qr + lm) * QS + c0;
      v16bf a = ld2(ap + half, ap + 16 + half);
      for (int t = 0; t < 4; ++t) {
        const unsigned short* bp = ks + (t * 16 + lm) * QS + c0 + half;
        v16bf bm = ld2(bp, bp + 16);
        s[t] = __builtin_amdgcn_wmma_f32_16x16x32_bf16(false, a, false, bm,
                                                       (short)0, s[t], false, false);
      }
    }

    // ---- online softmax (rows: VGPR idx a + lane-half) ----
    for (int a = 0; a < 8; ++a) {
      float mx = fmaxf(fmaxf(s[0][a], s[1][a]), fmaxf(s[2][a], s[3][a]));
      mx = grp_max(mx);
      float mn = fmaxf(m_st[a], mx);
      float sc = __expf(m_st[a] - mn);
      m_st[a] = mn;
      float rs = 0.f;
      for (int t = 0; t < 4; ++t) {
        float p = __expf(s[t][a] - mn);
        s[t][a] = p;
        rs += p;
      }
      rs = grp_sum(rs);
      l_st[a] = l_st[a] * sc + rs;
      for (int ct = 0; ct < 16; ++ct) o_acc[ct][a] *= sc;
    }

    // ---- stage P (C-layout f32 -> A-layout bf16 via wave-private LDS) ----
    unsigned short* pw = sm + P_E + w * 16 * VS;
    for (int t = 0; t < 4; ++t)
      for (int a = 0; a < 8; ++a)
        pw[(a + half) * VS + t * 16 + lm] = f2bf(s[t][a]);
    // same-wave LDS ops are in-order; compiler inserts s_wait_dscnt for RAW

    // ---- O += P V : 32x v_wmma_f32_16x16x32_bf16 ----
    for (int kt2 = 0; kt2 < 2; ++kt2) {
      int kb = kt2 * 32;
      const unsigned short* pp = pw + lm * VS + kb;
      v16bf pa = ld2(pp + half, pp + 16 + half);
      for (int ct = 0; ct < 16; ++ct) {
        const unsigned short* bp = vt + (ct * 16 + lm) * VS + kb + half;
        v16bf bm = ld2(bp, bp + 16);
        o_acc[ct] = __builtin_amdgcn_wmma_f32_16x16x32_bf16(false, pa, false, bm,
                                                            (short)0, o_acc[ct],
                                                            false, false);
      }
    }

    // ---- publish next tile: DMA complete + all waves done with cur ----
    if (w == 0) __builtin_amdgcn_s_wait_tensorcnt(0);
    __syncthreads();
  }

  // ---- epilogue: O/l + bias + residual, stored directly as [B,C,N] ----
  for (int a = 0; a < 8; ++a) {
    float inv = 1.0f / l_st[a];
    int n = q0 + qr + a + half;
    for (int ct = 0; ct < 16; ++ct) {
      int c = ct * 16 + lm;
      size_t idx = ((size_t)b * C_ + c) * N_ + n;
      out[idx] = o_acc[ct][a] * inv + bias[c] + x[idx];
    }
  }
}

// ---------------------------------------------------------------------------
extern "C" void kernel_launch(void* const* d_in, const int* in_sizes, int n_in,
                              void* d_out, int out_size, void* d_ws, size_t ws_size,
                              hipStream_t stream) {
  const float* x    = (const float*)d_in[0];   // [B, C, N]
  const float* W    = (const float*)d_in[1];   // [C, C]
  const float* bias = (const float*)d_in[2];   // [C]
  float* out = (float*)d_out;                  // [B, C, N]

  const size_t BNC = (size_t)B_ * N_ * C_;
  unsigned short* xn   = (unsigned short*)d_ws;  // xn   bf16 [B,N,C]
  unsigned short* xt   = xn + BNC;               // xt   bf16 [B,N,C]
  unsigned short* supT = xt + BNC;               // V^T  bf16 [B,C,N]
  unsigned short* Wt   = supT + BNC;             // W^T  bf16 [C,C]

  (void)hipFuncSetAttribute((const void*)gcn_attn,
                            hipFuncAttributeMaxDynamicSharedMemorySize, SMEM_BYTES);

  gcn_pack_w<<<dim3((C_ * C_) / 256), 256, 0, stream>>>(W, Wt);
  gcn_pack_x<<<dim3(N_ / 256, B_), 256, 0, stream>>>(x, xn, xt);
  gcn_support<<<dim3(C_ / 16, N_ / 128, B_), 256, 0, stream>>>(xt, Wt, supT);
  gcn_attn<<<dim3(N_ / MT, B_), 256, SMEM_BYTES, stream>>>(xn, supT, x, bias, out);
}
